// Postion_Att_75849122447945
// MI455X (gfx1250) — matrix operand order repaired and checked by reading, not compiled, not run on hardware
//
#include <hip/hip_runtime.h>

// Problem constants (match reference)
static constexpr int B_  = 16;
static constexpr int C_  = 512;
static constexpr int CQ_ = 256;   // C/2
static constexpr int N_  = 1024;  // W*H

typedef __attribute__((ext_vector_type(16))) __bf16        bf16x16;
typedef __attribute__((ext_vector_type(8)))  float         f32x8;
typedef __attribute__((ext_vector_type(4)))  unsigned int  u32x4;
typedef __attribute__((ext_vector_type(8)))  unsigned int  u32x8;

union BF16Frag {
    bf16x16      v;
    unsigned int w[8];
};

__device__ __forceinline__ unsigned short f2bf(float f) {
    unsigned int x = __builtin_bit_cast(unsigned int, f);
    unsigned int r = x + 0x7FFFu + ((x >> 16) & 1u);   // round-to-nearest-even
    return (unsigned short)(r >> 16);
}
__device__ __forceinline__ unsigned int pack_bf2(float lo, float hi) {
    return (unsigned int)f2bf(lo) | ((unsigned int)f2bf(hi) << 16);
}

// ---------------------------------------------------------------------------
// K0: one-shot f32 -> bf16 conversion of x (removes redundant per-tile VALU
// conversion work from the QKV GEMM). 4 elements per thread, float4 in,
// uint2 (4 packed bf16) out.
// ---------------------------------------------------------------------------
__global__ __launch_bounds__(256)
void cvt_x_bf16(const float* __restrict__ x, unsigned short* __restrict__ xbf)
{
    const size_t t = (size_t)blockIdx.x * 256 + threadIdx.x;
    const float4 f = ((const float4*)x)[t];
    uint2 o;
    o.x = pack_bf2(f.x, f.y);
    o.y = pack_bf2(f.z, f.w);
    ((uint2*)xbf)[t] = o;
}

// ---------------------------------------------------------------------------
// K1: out[o,n] = sum_c W[o,c] * xbf[b,c,n] + bias[o], bf16 output.
// One 16x16 tile per wave; K-loop over C in steps of 32 (v_wmma_f32_16x16x32_bf16).
// transposeOut=1 stores out as [N,O] (used for q so K2's A-frags are contiguous).
// ---------------------------------------------------------------------------
__global__ __launch_bounds__(256)
void qkv_gemm(const float* __restrict__ Wmat, const float* __restrict__ bias,
              const unsigned short* __restrict__ xbf, unsigned short* __restrict__ outw,
              int O, int transposeOut)
{
    const int b    = blockIdx.z;
    const int o0   = blockIdx.y * 16;
    const int lane = threadIdx.x & 31;
    const int wv   = threadIdx.x >> 5;
    const int n0   = blockIdx.x * 128 + wv * 16;
    const int hi   = lane >> 4;
    const int l    = lane & 15;

    const unsigned short* xb = xbf + (size_t)b * C_ * N_;

    f32x8 acc = {};
    for (int k0 = 0; k0 < C_; k0 += 32) {
        // A: W tile (16 o x 32 c), f32 -> bf16 pack (W is tiny; converted in-flight)
        BF16Frag a;
        const float* wrow = Wmat + (size_t)(o0 + l) * C_ + k0 + hi * 16;
#pragma unroll
        for (int v = 0; v < 8; ++v) {
            int kof = 2 * v;
            a.w[v] = pack_bf2(wrow[kof], wrow[kof + 1]);
        }
        // B: xbf tile (32 c x 16 n). Lane = col n0+l; low lanes K=0..15, hi lanes K=16..31.
        BF16Frag bb;
        const unsigned short* xc = xb + (size_t)(k0 + hi * 16) * N_ + (n0 + l);
#pragma unroll
        for (int v = 0; v < 8; ++v) {
            unsigned int lo16 = xc[(size_t)(2 * v) * N_];
            unsigned int h16  = xc[(size_t)(2 * v + 1) * N_];
            bb.w[v] = lo16 | (h16 << 16);
        }
        acc = __builtin_amdgcn_wmma_f32_16x16x32_bf16(false, a.v, false, bb.v,
                                                      (short)0, acc, false, false);
    }
    // D: lane -> col n0+l, vgpr r -> row o0 + hi*8 + r
    const int n = n0 + l;
    if (transposeOut) {
        unsigned short* ob = outw + (size_t)b * N_ * O + (size_t)n * O + (o0 + hi * 8);
#pragma unroll
        for (int r = 0; r < 8; ++r)
            ob[r] = f2bf(acc[r] + bias[o0 + hi * 8 + r]);
    } else {
        unsigned short* ob = outw + (size_t)b * O * N_;
#pragma unroll
        for (int r = 0; r < 8; ++r) {
            int orow = o0 + hi * 8 + r;
            ob[(size_t)orow * N_ + n] = f2bf(acc[r] + bias[orow]);
        }
    }
}

// ---------------------------------------------------------------------------
// K2: energy[i,j] = sum_c q[c,i]*k[c,j], then row softmax -> attn bf16 [N,N].
// q stored transposed [N,CQ] bf16 (contiguous A-frag dword loads).
// k slab (32 x 1024 bf16 = 64KB) is DMA'd into LDS once per K-step by the
// Tensor Data Mover (tensor_load_to_lds, 2D D#), synchronized with
// s_wait_tensorcnt + workgroup barriers; all 8 waves then pull B-fragments
// from LDS. The same 64KB LDS is re-used for the f32 energy block (union).
// ---------------------------------------------------------------------------
__global__ __launch_bounds__(256)
void attn_softmax(const unsigned short* __restrict__ qT,
                  const unsigned short* __restrict__ kk,
                  unsigned short* __restrict__ attn)
{
    __shared__ union {
        unsigned short ktile[32][N_];  // 64 KB: k slab for current K-step
        float          e[16][N_];      // 64 KB: energy block (after K-loop)
    } lds;

    const int b    = blockIdx.y;
    const int i0   = blockIdx.x * 16;
    const int lane = threadIdx.x & 31;
    const int wv   = threadIdx.x >> 5;
    const int hi   = lane >> 4;
    const int l    = lane & 15;

    const unsigned short* qb = qT + (size_t)b * N_ * CQ_;
    const unsigned short* kb = kk + (size_t)b * CQ_ * N_;

    const unsigned int lds_base =
        (unsigned int)(unsigned long long)(const void*)&lds.ktile[0][0];

    f32x8 acc[8];
#pragma unroll
    for (int t = 0; t < 8; ++t) acc[t] = (f32x8){};

    const int jbase = wv * 128;
    for (int k0 = 0; k0 < CQ_; k0 += 32) {
        __syncthreads();                       // previous slab fully consumed
        if (wv == 0) {
            // --- Tensor DMA descriptor (D#), 2D: 32 rows x 1024 cols of bf16 ---
            unsigned long long ga = (unsigned long long)(kb + (size_t)k0 * N_);
            u32x4 g0;
            g0.x = 1u;                                      // count=1, user mode
            g0.y = lds_base;                                // lds_addr
            g0.z = (unsigned int)ga;                        // global_addr[31:0]
            g0.w = (unsigned int)((ga >> 32) & 0x01FFFFFFull) | (2u << 30); // addr[56:32] | type=2
            u32x8 g1;
            g1.s0 = 1u << 16;                               // data_size = 2 bytes
            g1.s1 = (unsigned int)N_ << 16;                 // tensor_dim0[15:0]
            g1.s2 = (unsigned int)CQ_ << 16;                // tensor_dim0 hi | tensor_dim1[15:0]
            g1.s3 = (unsigned int)N_ << 16;                 // tensor_dim1 hi | tile_dim0
            g1.s4 = 32u;                                    // tile_dim1=32, tile_dim2=0
            g1.s5 = (unsigned int)N_;                       // tensor_dim0_stride[31:0]
            g1.s6 = 0u;
            g1.s7 = 0u;
            asm volatile("tensor_load_to_lds %0, %1" :: "s"(g0), "s"(g1) : "memory");
            __builtin_amdgcn_s_wait_tensorcnt(0);
        }
        __syncthreads();                       // slab visible to all waves

        // A: qT rows i0+l, contiguous bf16 pairs -> dword loads from global
        BF16Frag a;
        const unsigned short* qrow = qb + (size_t)(i0 + l) * CQ_ + k0 + hi * 16;
#pragma unroll
        for (int v = 0; v < 8; ++v) {
            int kof = 2 * v;
            a.w[v] = *(const unsigned int*)(qrow + kof);
        }
#pragma unroll
        for (int t = 0; t < 8; ++t) {
            BF16Frag bb;
            const int j = jbase + t * 16 + l;
#pragma unroll
            for (int v = 0; v < 8; ++v) {
                unsigned int lo16 = lds.ktile[hi * 16 + 2 * v][j];
                unsigned int h16  = lds.ktile[hi * 16 + 2 * v + 1][j];
                bb.w[v] = lo16 | (h16 << 16);
            }
            acc[t] = __builtin_amdgcn_wmma_f32_16x16x32_bf16(false, a.v, false, bb.v,
                                                             (short)0, acc[t], false, false);
        }
    }
    __syncthreads();   // all k-slab reads done; re-use LDS for energy

    // spill energy tiles to LDS: col = jbase + t*16 + l, row = hi*8 + r
#pragma unroll
    for (int t = 0; t < 8; ++t) {
        int j = jbase + t * 16 + l;
#pragma unroll
        for (int r = 0; r < 8; ++r) lds.e[hi * 8 + r][j] = acc[t][r];
    }
    __syncthreads();

    // softmax: wave wv handles rows 2wv and 2wv+1
#pragma unroll
    for (int rr = 0; rr < 2; ++rr) {
        const int row = wv * 2 + rr;
        float m = -3.0e38f;
        for (int jj = lane; jj < N_; jj += 32) m = fmaxf(m, lds.e[row][jj]);
#pragma unroll
        for (int off = 16; off > 0; off >>= 1) m = fmaxf(m, __shfl_xor(m, off, 32));
        float s = 0.f;
        for (int jj = lane; jj < N_; jj += 32) {
            float ev = __expf(lds.e[row][jj] - m);
            lds.e[row][jj] = ev;
            s += ev;
        }
#pragma unroll
        for (int off = 16; off > 0; off >>= 1) s += __shfl_xor(s, off, 32);
        const float inv = 1.0f / s;
        unsigned short* arow = attn + (size_t)b * N_ * N_ + (size_t)(i0 + row) * N_;
        for (int jj = lane; jj < N_; jj += 32) arow[jj] = f2bf(lds.e[row][jj] * inv);
    }
}

// ---------------------------------------------------------------------------
// K3: out[c,i] = sum_j v[c,j]*attn[i,j].  A = v tile (contiguous dword loads),
// B[k=j,n=i] = attn[i,j] (row-major attn -> contiguous dword loads per lane).
// global_prefetch pulls the next K-step's tiles while WMMAs drain.
// ---------------------------------------------------------------------------
__global__ __launch_bounds__(256)
void pv_gemm(const unsigned short* __restrict__ vv,
             const unsigned short* __restrict__ attn,
             float* __restrict__ outf)
{
    const int b    = blockIdx.z;
    const int c0   = blockIdx.y * 16;
    const int lane = threadIdx.x & 31;
    const int wv   = threadIdx.x >> 5;
    const int i0   = blockIdx.x * 128 + wv * 16;
    const int hi   = lane >> 4;
    const int l    = lane & 15;

    const unsigned short* vb = vv + (size_t)b * C_ * N_;
    const unsigned short* ab = attn + (size_t)b * N_ * N_;

    f32x8 acc = {};
    for (int j0 = 0; j0 < N_; j0 += 32) {
        BF16Frag a;
        const unsigned short* vrow = vb + (size_t)(c0 + l) * N_ + j0 + hi * 16;
        BF16Frag bb;
        const unsigned short* arow = ab + (size_t)(i0 + l) * N_ + j0 + hi * 16;
        // prefetch next K-step tiles (global_prefetch_b8)
        __builtin_prefetch((const void*)(vrow + 32), 0, 3);
        __builtin_prefetch((const void*)(arow + 32), 0, 3);
#pragma unroll
        for (int v = 0; v < 8; ++v) {
            int kof = 2 * v;
            a.w[v] = *(const unsigned int*)(vrow + kof);
        }
#pragma unroll
        for (int v = 0; v < 8; ++v)
            bb.w[v] = *(const unsigned int*)(arow + 2 * v);
        acc = __builtin_amdgcn_wmma_f32_16x16x32_bf16(false, a.v, false, bb.v,
                                                      (short)0, acc, false, false);
    }
    float* ob = outf + (size_t)b * C_ * N_;
    const int i = i0 + l;
#pragma unroll
    for (int r = 0; r < 8; ++r)
        ob[(size_t)(c0 + hi * 8 + r) * N_ + i] = acc[r];
}

// ---------------------------------------------------------------------------
// K4: per-pixel gating + residual. One wave per pixel: shuffle-reduced
// dot(Wa, out[:,i]), sigmoid gate, then two elementwise outputs.
// ---------------------------------------------------------------------------
__global__ __launch_bounds__(256)
void gate_residual(const float* __restrict__ outf, const float* __restrict__ x,
                   const float* __restrict__ Wa, const float* __restrict__ ba,
                   const float* __restrict__ gamma,
                   float* __restrict__ out_final, float* __restrict__ out_A)
{
    const int b    = blockIdx.y;
    const int lane = threadIdx.x & 31;
    const int wv   = threadIdx.x >> 5;
    const int i    = blockIdx.x * 8 + wv;

    const float* ob = outf + (size_t)b * C_ * N_ + i;
    const float* xb = x    + (size_t)b * C_ * N_ + i;

    float s = 0.f;
#pragma unroll 4
    for (int t = 0; t < C_ / 32; ++t) {
        int c = lane + t * 32;
        s += ob[(size_t)c * N_] * Wa[c];
    }
#pragma unroll
    for (int off = 16; off > 0; off >>= 1) s += __shfl_xor(s, off, 32);

    const float ag = 1.0f / (1.0f + __expf(-(s + ba[0])));
    const float g  = 1.0f - ag;
    const float gm = gamma[0];

    float* of = out_final + (size_t)b * C_ * N_ + i;
    float* oa = out_A     + (size_t)b * C_ * N_ + i;
#pragma unroll 4
    for (int t = 0; t < C_ / 32; ++t) {
        int c = lane + t * 32;
        float o = ob[(size_t)c * N_];
        of[(size_t)c * N_] = gm * o + xb[(size_t)c * N_];
        oa[(size_t)c * N_] = o * g;
    }
}

// ---------------------------------------------------------------------------
extern "C" void kernel_launch(void* const* d_in, const int* in_sizes, int n_in,
                              void* d_out, int out_size, void* d_ws, size_t ws_size,
                              hipStream_t stream)
{
    const float* x     = (const float*)d_in[0];
    const float* Wq    = (const float*)d_in[1];
    const float* bq    = (const float*)d_in[2];
    const float* Wk    = (const float*)d_in[3];
    const float* bk    = (const float*)d_in[4];
    const float* Wv    = (const float*)d_in[5];
    const float* bv    = (const float*)d_in[6];
    const float* Wa    = (const float*)d_in[7];
    const float* ba    = (const float*)d_in[8];
    const float* gamma = (const float*)d_in[9];
    float* outp = (float*)d_out;

    // Workspace layout (~113 MB):
    unsigned short* xbf  = (unsigned short*)d_ws;                    // [B][C][N]  bf16
    unsigned short* qT   = xbf  + (size_t)B_ * C_  * N_;             // [B][N][CQ] bf16 (q transposed)
    unsigned short* kbf  = qT   + (size_t)B_ * N_  * CQ_;            // [B][CQ][N] bf16
    unsigned short* vbf  = kbf  + (size_t)B_ * CQ_ * N_;             // [B][C][N]  bf16
    unsigned short* attn = vbf  + (size_t)B_ * C_  * N_;             // [B][N][N]  bf16
    float*          obuf = (float*)(attn + (size_t)B_ * N_ * N_);    // [B][C][N]  f32

    const dim3 blk(256);
    cvt_x_bf16<<<dim3((B_ * C_ * N_) / (4 * 256)), blk, 0, stream>>>(x, xbf);
    qkv_gemm<<<dim3(N_ / 128, CQ_ / 16, B_), blk, 0, stream>>>(Wq, bq, xbf, qT,  CQ_, 1);
    qkv_gemm<<<dim3(N_ / 128, CQ_ / 16, B_), blk, 0, stream>>>(Wk, bk, xbf, kbf, CQ_, 0);
    qkv_gemm<<<dim3(N_ / 128, C_  / 16, B_), blk, 0, stream>>>(Wv, bv, xbf, vbf, C_,  0);
    attn_softmax<<<dim3(N_ / 16, B_), blk, 0, stream>>>(qT, kbf, attn);
    pv_gemm<<<dim3(N_ / 128, C_ / 16, B_), blk, 0, stream>>>(vbf, attn, obuf);
    gate_residual<<<dim3(N_ / 8, B_), blk, 0, stream>>>(obuf, x, Wa, ba, gamma,
                                                        outp, outp + (size_t)B_ * C_ * N_);
}
